// EdgeFunction_34368328303087
// MI455X (gfx1250) — compile-verified, matrix operand order checked by reading, not run on hardware
//
#include <hip/hip_runtime.h>
#include <hip/hip_bf16.h>

typedef __attribute__((ext_vector_type(16))) _Float16 v16h;
typedef __attribute__((ext_vector_type(8)))  _Float16 v8h;
typedef __attribute__((ext_vector_type(8)))  float    v8f;

#define KB   4
#define NN   128
#define DIM  256
#define INFO 1024
#define ROWS 32
#define XS   1032   // padded LDS row stride in halfs (1024 + 8 -> 16B pad)

// Build a 16x32 f16 WMMA A/B fragment from a row-major [rows][K] surface.
// lane%16 selects the row (M for A, N for B); VGPR packing per ISA:
// VGPR0-3 = K kb..kb+7, VGPR4-7 = K kb+16..kb+23, kb = (lane/16)*8.
__device__ __forceinline__ v16h frag_from(const _Float16* p) {
  union { v16h v; v8h h[2]; } f;
  f.h[0] = *(const v8h*)p;          // K kb .. kb+7
  f.h[1] = *(const v8h*)(p + 16);   // K kb+16 .. kb+23
  return f.v;
}

// One-time: convert + transpose weights to f16 so B-fragments are two
// contiguous b128 loads. w1t = [1024 N][1024 K], w2t = [256 N][1024 K].
__global__ void prep_weights(const float* __restrict__ W1,
                             const float* __restrict__ W2,
                             _Float16* __restrict__ w1t,
                             _Float16* __restrict__ w2t) {
  int idx = blockIdx.x * 256 + threadIdx.x;
  if (idx < (1 << 20)) {                       // W1: 1024x1024
    int nn = idx >> 10, kk = idx & 1023;
    w1t[idx] = (_Float16)W1[kk * 1024 + nn];
  } else if (idx < (1 << 20) + 256 * 1024) {   // W2: 1024x256
    int j = idx - (1 << 20);
    int nn = j >> 10, kk = j & 1023;
    w2t[j] = (_Float16)W2[kk * 256 + nn];
  }
}

__global__ __launch_bounds__(256) void edge_mlp_kernel(
    const float* __restrict__ u,  const float* __restrict__ h_v,
    const float* __restrict__ h_e, const float* __restrict__ V,
    const int* __restrict__ E,    const float* __restrict__ b1,
    const float* __restrict__ b2, const _Float16* __restrict__ w1t,
    const _Float16* __restrict__ w2t, float* __restrict__ out) {
  __shared__ _Float16 Xs[ROWS * XS];   // edge features, f16
  __shared__ _Float16 Hs[ROWS * XS];   // hidden after elu, f16

  const int bid = blockIdx.x;
  const int mt  = bid & 3;            // 4 m-tiles of 32
  const int n   = (bid >> 2) & 127;
  const int k   = bid >> 9;
  const int m0  = mt * ROWS;
  const int tid = threadIdx.x;

  // ---- Prologue: build X tile [32 x 1024] = [src | tgt | glb | h_e] in f16
  const float vn = V[k * NN + n];
  for (int e = tid; e < ROWS * INFO; e += 256) {
    int lr = e >> 10;
    int c  = e & 1023;
    int m  = m0 + lr;
    float val;
    if (c < 256)       val = h_v[(k * NN + n) * DIM + c] * vn;
    else if (c < 512)  val = h_v[(k * NN + m) * DIM + (c - 256)] * V[k * NN + m];
    else if (c < 768)  val = u[k * DIM + (c - 512)];
    else               val = h_e[(((long)(k * NN + n)) * NN + m) * DIM + (c - 768)];
    Xs[lr * XS + c] = (_Float16)val;
  }
  __syncthreads();

  const int lane  = tid & 31;
  const int wv    = tid >> 5;       // 8 waves
  const int col16 = lane & 15;
  const int kb    = (lane >> 4) * 8;
  const int rhalf = (lane >> 4) << 3;   // +0 or +8 row offset inside a tile
  const v8f zero = {};

  // ---- GEMM1: H = elu(X @ W1 + b1)
  // Wave w owns col-tiles ct0..ct0+7 and BOTH row-tiles, so each B fragment
  // (global/L2) feeds two WMMAs; A fragments come from LDS.
  {
    const int ct0 = wv * 8;
    v8f acc[2][8];
#pragma unroll
    for (int r = 0; r < 2; ++r)
#pragma unroll
      for (int i = 0; i < 8; ++i) acc[r][i] = zero;
    const _Float16* arow0 = &Xs[(col16) * XS + kb];
    const _Float16* arow1 = &Xs[(16 + col16) * XS + kb];
    for (int ks = 0; ks < 32; ++ks) {
      v16h a0 = frag_from(arow0 + ks * 32);
      v16h a1 = frag_from(arow1 + ks * 32);
      const _Float16* bbase = w1t + (ct0 * 16 + col16) * INFO + ks * 32 + kb;
#pragma unroll
      for (int i = 0; i < 8; ++i) {
        v16h b = frag_from(bbase + i * 16 * INFO);
        acc[0][i] = __builtin_amdgcn_wmma_f32_16x16x32_f16(
            false, a0, false, b, (short)0, acc[0][i], false, false);
        acc[1][i] = __builtin_amdgcn_wmma_f32_16x16x32_f16(
            false, a1, false, b, (short)0, acc[1][i], false, false);
      }
    }
#pragma unroll
    for (int r = 0; r < 2; ++r) {
      int rbase = r * 16 + rhalf;
#pragma unroll
      for (int i = 0; i < 8; ++i) {
        int ncol   = (ct0 + i) * 16 + col16;
        float bias = b1[ncol];
#pragma unroll
        for (int j = 0; j < 8; ++j) {
          float x = acc[r][i][j] + bias;
          x = x > 0.f ? x : (__expf(x) - 1.f);        // elu
          Hs[(rbase + j) * XS + ncol] = (_Float16)x;
        }
      }
    }
  }
  __syncthreads();

  // ---- GEMM2: out = (E-mask ? H @ W2 + b2 : 0) + h_e
  {
    const int ct0 = wv * 2;           // 8 waves x 2 = 16 col-tiles (256 cols)
    v8f acc[2][2];
#pragma unroll
    for (int r = 0; r < 2; ++r)
#pragma unroll
      for (int i = 0; i < 2; ++i) acc[r][i] = zero;
    const _Float16* arow0 = &Hs[(col16) * XS + kb];
    const _Float16* arow1 = &Hs[(16 + col16) * XS + kb];
    for (int ks = 0; ks < 32; ++ks) {
      v16h a0 = frag_from(arow0 + ks * 32);
      v16h a1 = frag_from(arow1 + ks * 32);
      const _Float16* bbase = w2t + (ct0 * 16 + col16) * INFO + ks * 32 + kb;
#pragma unroll
      for (int i = 0; i < 2; ++i) {
        v16h b = frag_from(bbase + i * 16 * INFO);
        acc[0][i] = __builtin_amdgcn_wmma_f32_16x16x32_f16(
            false, a0, false, b, (short)0, acc[0][i], false, false);
        acc[1][i] = __builtin_amdgcn_wmma_f32_16x16x32_f16(
            false, a1, false, b, (short)0, acc[1][i], false, false);
      }
    }
#pragma unroll
    for (int r = 0; r < 2; ++r) {
      int rbase = r * 16 + rhalf;
#pragma unroll
      for (int i = 0; i < 2; ++i) {
        int ncol   = (ct0 + i) * 16 + col16;
        float bias = b2[ncol];
#pragma unroll
        for (int j = 0; j < 8; ++j) {
          int  lr   = rbase + j;
          int  m    = m0 + lr;
          long eidx = ((long)(k * NN + n)) * NN + m;
          bool msk  = (E[eidx * 2 + 1] == 1);
          float he  = h_e[eidx * DIM + ncol];
          out[eidx * DIM + ncol] = (msk ? (acc[r][i][j] + bias) : 0.f) + he;
        }
      }
    }
  }
}

extern "C" void kernel_launch(void* const* d_in, const int* in_sizes, int n_in,
                              void* d_out, int out_size, void* d_ws, size_t ws_size,
                              hipStream_t stream) {
  const float* u   = (const float*)d_in[0];
  const float* h_v = (const float*)d_in[1];
  const float* h_e = (const float*)d_in[2];
  const float* V   = (const float*)d_in[3];
  const int*   E   = (const int*)d_in[4];
  const float* W1  = (const float*)d_in[5];
  const float* b1  = (const float*)d_in[6];
  const float* W2  = (const float*)d_in[7];
  const float* b2  = (const float*)d_in[8];

  _Float16* w1t = (_Float16*)d_ws;                                    // 2 MB
  _Float16* w2t = (_Float16*)((char*)d_ws + (size_t)1024 * 1024 * 2); // 0.5 MB

  int total = (1 << 20) + 256 * 1024;
  prep_weights<<<(total + 255) / 256, 256, 0, stream>>>(W1, W2, w1t, w2t);

  // 4 batches * 128 src-nodes * 4 m-tiles = 2048 workgroups of 256 threads
  edge_mlp_kernel<<<KB * NN * (NN / ROWS), 256, 0, stream>>>(
      u, h_v, h_e, V, E, b1, b2, w1t, w2t, (float*)d_out);
}
